// GroupCNN_34333968564689
// MI455X (gfx1250) — matrix operand order repaired and checked by reading, not compiled, not run on hardware
//
#include <hip/hip_runtime.h>
#include <hip/hip_bf16.h>

typedef _Float16 f16;
typedef __attribute__((ext_vector_type(16))) _Float16 v16h;
typedef __attribute__((ext_vector_type(8)))  _Float16 v8h;
typedef __attribute__((ext_vector_type(8)))  float    v8f;

#define CDIV(a,b) (((a)+(b)-1)/(b))

// ---------------------------------------------------------------------------
// rot90 applied r times (counter-clockwise, numpy semantics) for 3x3 kernels:
// rotated[i][j] = w[si][sj] with (si,sj) = rotmap^r(i,j), one step: (i,j)->(j,2-i)
__device__ __forceinline__ void rotmap(int r, int i, int j, int& si, int& sj) {
    si = i; sj = j;
    #pragma unroll
    for (int t = 0; t < 3; ++t) {
        if (t < r) { int tmp = si; si = sj; sj = 2 - tmp; }
    }
}

// ---------------------------------------------------------------------------
// Weight prep: p4conv_z2.  w1:(16,1,3,3) f32 -> A[64][32] f16, M = o*4+r,
// K = kh*3+kw (9 valid, zero-padded to 32).
__global__ void prep_z2(const float* __restrict__ w, f16* __restrict__ A) {
    int idx = blockIdx.x * blockDim.x + threadIdx.x;
    if (idx >= 64 * 32) return;
    int m = idx / 32, k = idx % 32;
    int o = m >> 2, r = m & 3;
    f16 val = (f16)0.f;
    if (k < 9) {
        int kh = k / 3, kw = k % 3, si, sj;
        rotmap(r, kh, kw, si, sj);
        val = (f16)w[o * 9 + si * 3 + sj];
    }
    A[idx] = val;
}

// Weight prep: p4conv_p4.  w:(O,I,4,3,3) f32 -> A[O*4][KPAD] f16.
// M = o*4+r, K = (i*4+s)*9 + kh*3 + kw; A[m][k] = w[o,i,(s-r)%4, rotmap(r,kh,kw)]
__global__ void prep_p4(const float* __restrict__ w, f16* __restrict__ A,
                        int O, int I, int KPAD) {
    int idx = blockIdx.x * blockDim.x + threadIdx.x;
    if (idx >= O * 4 * KPAD) return;
    int m = idx / KPAD, k = idx % KPAD;
    int o = m >> 2, r = m & 3;
    f16 val = (f16)0.f;
    if (k < I * 4 * 9) {
        int c = k / 9, rr = k % 9, kh = rr / 3, kw = rr % 3;
        int i = c >> 2, s = c & 3;
        int s0 = (s - r + 4) & 3;
        int si, sj;
        rotmap(r, kh, kw, si, sj);
        val = (f16)w[(((o * I + i) * 4 + s0) * 9) + si * 3 + sj];
    }
    A[idx] = val;
}

// Weight prep: fc1 (32,800) f32 -> A[32][832] f16 zero-padded.
__global__ void prep_fc(const float* __restrict__ w, f16* __restrict__ A) {
    int idx = blockIdx.x * blockDim.x + threadIdx.x;
    if (idx >= 32 * 832) return;
    int m = idx / 832, k = idx % 832;
    A[idx] = (k < 800) ? (f16)w[m * 800 + k] : (f16)0.f;
}

// f32 -> f16 activation conversion
__global__ void cvt_f32_f16(const float* __restrict__ in, f16* __restrict__ out, long n) {
    long i = (long)blockIdx.x * blockDim.x + threadIdx.x;
    if (i < n) out[i] = (f16)in[i];
}

// 2x2 spatial max pool over NCHW f16
__global__ void maxpool2(const f16* __restrict__ in, f16* __restrict__ out,
                         int C, int H, int W, long total) {
    long i = (long)blockIdx.x * blockDim.x + threadIdx.x;
    if (i >= total) return;
    int Wo = W >> 1, Ho = H >> 1;
    int ow = (int)(i % Wo); long t = i / Wo;
    int oh = (int)(t % Ho); t /= Ho;
    int c  = (int)(t % C);  long b = t / C;
    const f16* p = &in[(((b * C) + c) * H + oh * 2) * (size_t)W + ow * 2];
    f16 a0 = p[0], a1 = p[1], a2 = p[W], a3 = p[W + 1];
    f16 m0 = a0 > a1 ? a0 : a1;
    f16 m1 = a2 > a3 ? a2 : a3;
    out[i] = m0 > m1 ? m0 : m1;
}

// ---------------------------------------------------------------------------
// Implicit-GEMM conv via WMMA with cooperative LDS staging.
// Workgroup = (COUT/16) waves; wave w owns M-tile w; the block shares one
// 32(K)x32(N) im2col tile staged in double-buffered LDS.  Each wave issues
// two v_wmma_f32_16x16x32_f16 per K-chunk (N sub-tiles 0..15 / 16..31).
// B-fragment layout used here (lane half h holds K = 16h..16h+15 contiguously)
// maps to two ds_load_b128 per fragment from LDS rows of 32 f16.
template<int CIN, int H, int W, int HO, int WO, int COUT, int KTOT, int KPAD>
__launch_bounds__(COUT * 2)
__global__ void conv_wmma(const f16* __restrict__ act_in, const f16* __restrict__ Aw,
                          const float* __restrict__ bias, f16* __restrict__ act_out,
                          int Bn) {
    constexpr int MT     = COUT / 16;        // waves per block
    constexpr int ROWSPT = MT;               // n_local stride while staging
    constexpr int E      = (32 + ROWSPT - 1) / ROWSPT;  // staging slots/thread
    constexpr int NCH    = KPAD / 32;        // K chunks

    __shared__ f16 lbs[2][32 * 32];          // [buf][n_local][kk]  (2 x 2KB)

    const int tid  = threadIdx.x;
    const int wave = tid >> 5;               // == M-tile index
    const int lane = tid & 31;
    const int half = lane >> 4;
    const int lid  = lane & 15;
    const int ntile = blockIdx.x;            // 32-wide N block
    const int Ntot  = Bn * HO * WO;

    // ---- staging slots: fixed kk column per thread, hoisted n decode ----
    const int kk = lane;                     // K offset within chunk (0..31)
    size_t sbase[E];
    bool   svalid[E];
    #pragma unroll
    for (int s = 0; s < E; ++s) {
        int n_local = wave + s * ROWSPT;
        int n = ntile * 32 + n_local;
        bool ok = (n_local < 32) && (n < Ntot);
        svalid[s] = ok;
        size_t base = 0;
        if (ok) {
            int b  = n / (HO * WO);
            int rm = n % (HO * WO);
            int oh = rm / WO, ow = rm % WO;
            base = ((size_t)b * CIN * H + oh) * W + ow;
        }
        sbase[s] = base;
    }

    auto stage = [&](int buf, int kc) {
        int k = kc + kk;
        bool kv = (k < KTOT);
        size_t koff = 0;
        if (kv) {
            int c = k / 9, rr = k % 9, kh = rr / 3, kwd = rr % 3;
            koff = ((size_t)c * H + kh) * W + kwd;
        }
        #pragma unroll
        for (int s = 0; s < E; ++s) {
            int n_local = wave + s * ROWSPT;
            if (n_local < 32) {
                f16 v = (f16)0.f;
                if (kv && svalid[s]) v = act_in[sbase[s] + koff];
                lbs[buf][n_local * 32 + kk] = v;
            }
        }
    };

    // ---- A row pointer (both halves cover all 16 M rows, different K) ----
    const f16* arow = Aw + (size_t)(wave * 16 + lid) * KPAD;

    v8f acc0 = {}, acc1 = {};
    stage(0, 0);
    for (int i = 0; i < NCH; ++i) {
        __syncthreads();
        if (i + 1 < NCH) stage((i + 1) & 1, (i + 1) * 32);
        const int kc  = i * 32;
        const int buf = i & 1;
        if (i + 1 < NCH) __builtin_prefetch(arow + kc + 32, 0, 1);

        // A fragment: two contiguous 8xf16 global loads, K = half*16..+15 (same order as B)
        v8h lo = *(const v8h*)(arow + kc + half * 16);
        v8h hi = *(const v8h*)(arow + kc + half * 16 + 8);
        v16h afrag;
        #pragma unroll
        for (int t = 0; t < 8; ++t) { afrag[t] = lo[t]; afrag[8 + t] = hi[t]; }

        // B fragments from LDS: lane covers N=n_local, K=half*16..+15 contiguous
        v16h b0 = *(const v16h*)&lbs[buf][lid * 32 + half * 16];
        v16h b1 = *(const v16h*)&lbs[buf][(16 + lid) * 32 + half * 16];

        acc0 = __builtin_amdgcn_wmma_f32_16x16x32_f16(
            false, afrag, false, b0, (short)0, acc0, false, false);
        acc1 = __builtin_amdgcn_wmma_f32_16x16x32_f16(
            false, afrag, false, b1, (short)0, acc1, false, false);
    }

    // ---- epilogue: C/D layout VGPR j -> M = j + 8*half, N = lid ----
    int n0 = ntile * 32 + lid;
    int n1 = n0 + 16;
    int b0i = 0, oh0 = 0, ow0 = 0, b1i = 0, oh1 = 0, ow1 = 0;
    bool v0 = (n0 < Ntot), v1 = (n1 < Ntot);
    if (v0) { b0i = n0 / (HO * WO); int rm = n0 % (HO * WO); oh0 = rm / WO; ow0 = rm % WO; }
    if (v1) { b1i = n1 / (HO * WO); int rm = n1 % (HO * WO); oh1 = rm / WO; ow1 = rm % WO; }
    #pragma unroll
    for (int j = 0; j < 8; ++j) {
        int m = wave * 16 + j + half * 8;
        float bv = bias[m >> 2];                  // per rotation-group (o = m/4)
        if (v0) {
            float v = acc0[j] + bv; v = v > 0.f ? v : 0.f;
            act_out[(((size_t)b0i * COUT + m) * HO + oh0) * WO + ow0] = (f16)v;
        }
        if (v1) {
            float v = acc1[j] + bv; v = v > 0.f ? v : 0.f;
            act_out[(((size_t)b1i * COUT + m) * HO + oh1) * WO + ow1] = (f16)v;
        }
    }
}

// fc1: (Bn,800) f16 x (32,800)^T -> ReLU -> (Bn,32) f16, WMMA path.
__launch_bounds__(32)
__global__ void fc1_wmma(const f16* __restrict__ in, const f16* __restrict__ Aw,
                         const float* __restrict__ bias, f16* __restrict__ out, int Bn) {
    const int KTOT = 800, KPAD = 832;
    const int ntile = blockIdx.x, mtile = blockIdx.y;
    const int lane = threadIdx.x, half = lane >> 4, lid = lane & 15;

    int n = ntile * 16 + lid;
    const bool nvalid = (n < Bn);
    if (!nvalid) n = 0;
    const f16* arow = Aw + (size_t)(mtile * 16 + lid) * KPAD;
    const f16* brow = in + (size_t)n * KTOT;

    v8f acc = {};
    for (int kc = 0; kc < KPAD; kc += 32) {
        if (kc + 32 < KPAD) __builtin_prefetch(brow + kc + 32, 0, 1);
        v16h afrag, bfrag;
        #pragma unroll
        for (int v = 0; v < 8; ++v) {
            int k = kc + ((v < 4) ? 2 * v : 2 * v + 8) + (half ? 8 : 0);
            afrag[2 * v]     = arow[k];
            afrag[2 * v + 1] = arow[k + 1];
        }
        #pragma unroll
        for (int v = 0; v < 8; ++v) {
            int k0 = kc + ((v < 4) ? 2 * v : 2 * v + 8) + (half ? 8 : 0);
            bfrag[2 * v]     = (k0     < KTOT) ? brow[k0]     : (f16)0.f;
            bfrag[2 * v + 1] = (k0 + 1 < KTOT) ? brow[k0 + 1] : (f16)0.f;
        }
        acc = __builtin_amdgcn_wmma_f32_16x16x32_f16(
            false, afrag, false, bfrag, (short)0, acc, false, false);
    }
    if (!nvalid) return;
    #pragma unroll
    for (int j = 0; j < 8; ++j) {
        int m = mtile * 16 + j + half * 8;
        float v = acc[j] + bias[m];
        out[(size_t)n * 32 + m] = (f16)(v > 0.f ? v : 0.f);
    }
}

// fc2 + log_softmax, one thread per row (tiny: 1024x10x32)
__global__ void fc2_lsm(const f16* __restrict__ h, const float* __restrict__ w,
                        const float* __restrict__ bias, float* __restrict__ out, int Bn) {
    int t = blockIdx.x * blockDim.x + threadIdx.x;
    if (t >= Bn) return;
    float hv[32];
    #pragma unroll
    for (int k = 0; k < 32; ++k) hv[k] = (float)h[(size_t)t * 32 + k];
    float lg[10], mx = -1e30f;
    #pragma unroll
    for (int o = 0; o < 10; ++o) {
        float s = bias[o];
        #pragma unroll
        for (int k = 0; k < 32; ++k) s += w[o * 32 + k] * hv[k];
        lg[o] = s; mx = s > mx ? s : mx;
    }
    float sum = 0.f;
    #pragma unroll
    for (int o = 0; o < 10; ++o) sum += __expf(lg[o] - mx);
    float lse = __logf(sum);
    #pragma unroll
    for (int o = 0; o < 10; ++o) out[(size_t)t * 10 + o] = lg[o] - mx - lse;
}

// ---------------------------------------------------------------------------
extern "C" void kernel_launch(void* const* d_in, const int* in_sizes, int n_in,
                              void* d_out, int out_size, void* d_ws, size_t ws_size,
                              hipStream_t stream) {
    const float* x    = (const float*)d_in[0];
    const float* w1   = (const float*)d_in[1];
    const float* b1   = (const float*)d_in[2];
    const float* w2   = (const float*)d_in[3];
    const float* b2   = (const float*)d_in[4];
    const float* w3   = (const float*)d_in[5];
    const float* b3   = (const float*)d_in[6];
    const float* w4   = (const float*)d_in[7];
    const float* b4   = (const float*)d_in[8];
    const float* fc1w = (const float*)d_in[9];
    const float* fc1b = (const float*)d_in[10];
    const float* fc2w = (const float*)d_in[11];
    const float* fc2b = (const float*)d_in[12];
    float* out = (float*)d_out;

    const int Bn = in_sizes[0] / (32 * 32);   // 1024

    f16* ws = (f16*)d_ws;
    size_t off = 0;
    auto alloc = [&](size_t e) { size_t o = off; off += (e + 127) & ~(size_t)127; return o; };
    const size_t sS  = alloc((size_t)Bn * 32 * 32);        // x in f16
    const size_t sA  = alloc((size_t)Bn * 64 * 30 * 30);   // ping
    const size_t sB  = alloc((size_t)Bn * 128 * 28 * 28);  // pong
    const size_t sA1 = alloc(64  * 32);
    const size_t sA2 = alloc(128 * 576);
    const size_t sA3 = alloc(48  * 1152);
    const size_t sA4 = alloc(32  * 448);
    const size_t sAf = alloc(32  * 832);
    (void)ws_size;

    // input f32 -> f16
    {
        long n = (long)Bn * 32 * 32;
        cvt_f32_f16<<<CDIV(n, 256), 256, 0, stream>>>(x, ws + sS, n);
    }
    // rotated / rolled / K-padded weight matrices
    prep_z2<<<CDIV(64 * 32, 256), 256, 0, stream>>>(w1, ws + sA1);
    prep_p4<<<CDIV(128 * 576, 256), 256, 0, stream>>>(w2, ws + sA2, 32, 16, 576);
    prep_p4<<<CDIV(48 * 1152, 256), 256, 0, stream>>>(w3, ws + sA3, 12, 32, 1152);
    prep_p4<<<CDIV(32 * 448, 256), 256, 0, stream>>>(w4, ws + sA4, 8, 12, 448);
    prep_fc<<<CDIV(32 * 832, 256), 256, 0, stream>>>(fc1w, ws + sAf);

    // conv1: (Bn,1,32,32) -> (Bn,64,30,30)    block = 4 waves
    conv_wmma<1, 32, 32, 30, 30, 64, 9, 32>
        <<<CDIV(Bn * 30 * 30, 32), 64 * 2, 0, stream>>>(ws + sS, ws + sA1, b1, ws + sA, Bn);
    // conv2: (Bn,64,30,30) -> (Bn,128,28,28)  block = 8 waves
    conv_wmma<64, 30, 30, 28, 28, 128, 576, 576>
        <<<CDIV(Bn * 28 * 28, 32), 128 * 2, 0, stream>>>(ws + sA, ws + sA2, b2, ws + sB, Bn);
    // pool1 -> (Bn,128,14,14)
    {
        long n = (long)Bn * 128 * 14 * 14;
        maxpool2<<<CDIV(n, 256), 256, 0, stream>>>(ws + sB, ws + sA, 128, 28, 28, n);
    }
    // conv3: (Bn,128,14,14) -> (Bn,48,12,12)  block = 3 waves
    conv_wmma<128, 14, 14, 12, 12, 48, 1152, 1152>
        <<<CDIV(Bn * 12 * 12, 32), 48 * 2, 0, stream>>>(ws + sA, ws + sA3, b3, ws + sB, Bn);
    // conv4: (Bn,48,12,12) -> (Bn,32,10,10)   block = 2 waves
    conv_wmma<48, 12, 12, 10, 10, 32, 432, 448>
        <<<CDIV(Bn * 10 * 10, 32), 32 * 2, 0, stream>>>(ws + sB, ws + sA4, b4, ws + sA, Bn);
    // pool2 -> (Bn,32,5,5) == flattened (Bn,800)
    {
        long n = (long)Bn * 32 * 5 * 5;
        maxpool2<<<CDIV(n, 256), 256, 0, stream>>>(ws + sA, ws + sB, 32, 10, 10, n);
    }
    // fc1 -> (Bn,32) f16 (ReLU)
    fc1_wmma<<<dim3(CDIV(Bn, 16), 2), 32, 0, stream>>>(ws + sB, ws + sAf, fc1b, ws + sA, Bn);
    // fc2 + log_softmax -> d_out (Bn,10) f32
    fc2_lsm<<<CDIV(Bn, 256), 256, 0, stream>>>(ws + sA, fc2w, fc2b, out, Bn);
}
